// ResidualVQLayer_52441550684350
// MI455X (gfx1250) — compile-verified, matrix operand order checked by reading, not run on hardware
//
#include <hip/hip_runtime.h>
#include <stdint.h>

typedef __attribute__((ext_vector_type(16))) __bf16 v16bf;
typedef __attribute__((ext_vector_type(8)))  float  v8f;

#define TOK_TOTAL  65536
#define DIN        1024
#define DRVQ       64
#define NQ         4
#define CODES      32
#define BLOCK_T    128
#define KC         128
#define NC         128

// LDS layout (bytes), phase-overlapped:
//  [OFF_W ] 16896: Phase A: W_in chunk transposed [64][132] bf16 ; Phase D: W_out chunk transposed [128][66] bf16
//  [OFF_XQ] 16896: Phase A: per-wave x stage [16][132] bf16      ; Phase C/D: per-wave quant [16][68] bf16
//  [OFF_R ] 17408: per-wave residual [16][68] f32
//  [OFF_CB]  8320: codebook [32][65] f32 (padded stride -> conflict-free)
#define OFF_W    0
#define OFF_XQ   16896
#define OFF_R    33792
#define OFF_CB   51200
#define SMEM_TOT 59520

__device__ __forceinline__ uint16_t f2bf(float f) {
  uint32_t u = __float_as_uint(f);
  u += 0x7FFFu + ((u >> 16) & 1u);          // round-to-nearest-even
  return (uint16_t)(u >> 16);
}
__device__ __forceinline__ uint32_t pk2bf(float lo, float hi) {
  return (uint32_t)f2bf(lo) | ((uint32_t)f2bf(hi) << 16);
}
// 16-bit A/B fragment K striping (ISA 7.12.2): VGPR v, lane-half hi
__device__ __forceinline__ int kbase(int v, int hi) {
  return ((v & 4) << 2) + (hi << 3) + ((v & 3) << 1);
}

union Frag { uint32_t u[8]; v16bf v; };

__global__ void rvq_zero(float* p) { *p = 0.f; }

__global__ void rvq_finish(const float* p, float* loss) {
  *loss = *p * (1.0f / (float)(TOK_TOTAL * DRVQ));
}

__global__ __launch_bounds__(BLOCK_T) void rvq_fused(
    const float* __restrict__ x,    const float* __restrict__ Win,
    const float* __restrict__ bin,  const float* __restrict__ cbs,
    const float* __restrict__ Wout, const float* __restrict__ bout,
    float* __restrict__ zq, float* __restrict__ idxf, float* __restrict__ lacc)
{
  __shared__ __align__(16) unsigned char smem[SMEM_TOT];
  uint16_t* wT    = (uint16_t*)(smem + OFF_W);
  uint16_t* xsAll = (uint16_t*)(smem + OFF_XQ);
  float*    rsAll = (float*)   (smem + OFF_R);
  float*    cbS   = (float*)   (smem + OFF_CB);

  const int tid  = threadIdx.x;
  const int wave = tid >> 5;
  const int lane = tid & 31;
  const int m    = lane & 15;     // row (A) / col (B,C) within 16
  const int hi   = lane >> 4;     // lane half
  const size_t wtok0 = (size_t)blockIdx.x * 64 + (size_t)wave * 16;

  uint16_t* xs = xsAll + wave * (16 * 132);   // phase A stage
  uint16_t* qs = xsAll + wave * (16 * 68);    // phase C quant (region reused)
  float*    rs = rsAll + wave * (16 * 68);    // residual

  v8f acc[4] = {};  // x_proj tile: 16 tokens x 64 dims

  // ---------------- Phase A: x_proj = x @ W_in (bf16 WMMA, K chunks of 128)
  for (int kc = 0; kc < DIN; kc += KC) {
    // W_in chunk -> LDS, transposed bf16: wT[n*132 + k]
    #pragma unroll
    for (int it = 0; it < 16; ++it) {
      int e = (it * BLOCK_T + tid) << 2;          // 8192 f32 per chunk
      int k = e >> 6, n = e & 63;
      const float4 w4 = *(const float4*)(Win + (size_t)(kc + k) * DRVQ + n);
      wT[(n + 0) * 132 + k] = f2bf(w4.x);
      wT[(n + 1) * 132 + k] = f2bf(w4.y);
      wT[(n + 2) * 132 + k] = f2bf(w4.z);
      wT[(n + 3) * 132 + k] = f2bf(w4.w);
    }
    // x tile (16 rows x 128) -> LDS bf16 per wave: xs[r*132 + c]
    #pragma unroll
    for (int it = 0; it < 16; ++it) {
      int e = (it * 32 + lane) << 2;              // 2048 f32 per wave
      int r = e >> 7, c = e & 127;
      const float4 x4 = *(const float4*)(x + (wtok0 + r) * DIN + kc + c);
      *(uint32_t*)(xs + r * 132 + c)     = pk2bf(x4.x, x4.y);
      *(uint32_t*)(xs + r * 132 + c + 2) = pk2bf(x4.z, x4.w);
    }
    __syncthreads();
    #pragma unroll
    for (int kk = 0; kk < KC; kk += 32) {
      Frag a;
      #pragma unroll
      for (int v = 0; v < 8; ++v)
        a.u[v] = *(const uint32_t*)(xs + m * 132 + kk + kbase(v, hi));
      #pragma unroll
      for (int nt = 0; nt < 4; ++nt) {
        Frag b;
        #pragma unroll
        for (int v = 0; v < 8; ++v)
          b.u[v] = *(const uint32_t*)(wT + (nt * 16 + m) * 132 + kk + kbase(v, hi));
        acc[nt] = __builtin_amdgcn_wmma_f32_16x16x32_bf16(
            false, a.v, false, b.v, (short)0, acc[nt], false, false);
      }
    }
    __syncthreads();
  }
  // + b_in ; residual := x_proj (LDS f32, stride 68 -> conflict-free writes)
  #pragma unroll
  for (int nt = 0; nt < 4; ++nt) {
    const float bi = bin[nt * 16 + m];
    #pragma unroll
    for (int r = 0; r < 8; ++r) {
      float v = acc[nt][r] + bi;
      acc[nt][r] = v;
      rs[(hi * 8 + r) * 68 + nt * 16 + m] = v;
    }
  }

  // ---------------- Phase B: 4-round residual VQ (one code per lane)
  float lsum = 0.f;
  for (int q = 0; q < NQ; ++q) {
    __syncthreads();
    #pragma unroll
    for (int it = 0; it < 16; ++it) {
      int e = it * BLOCK_T + tid;                 // 2048 f32 codebook
      cbS[(e >> 6) * 65 + (e & 63)] = cbs[(size_t)q * (CODES * DRVQ) + e];
    }
    __syncthreads();
    for (int t = 0; t < 16; ++t) {
      float* rrow = rs + t * 68;
      const float* crow = cbS + lane * 65;
      float dist = 0.f;
      #pragma unroll 4
      for (int d = 0; d < DRVQ; ++d) {            // rrow[d] broadcast, crow conflict-free
        float df = rrow[d] - crow[d];
        dist = fmaf(df, df, dist);
      }
      int best = lane;
      #pragma unroll
      for (int s = 16; s >= 1; s >>= 1) {         // wave32 argmin, ties -> lowest idx
        float od = __shfl_xor(dist, s, 32);
        int   ob = __shfl_xor(best, s, 32);
        if (od < dist || (od == dist && ob < best)) { dist = od; best = ob; }
      }
      if (lane == 0)
        idxf[(size_t)q * TOK_TOTAL + wtok0 + t] = (float)best;
      const float* qrow = cbS + best * 65;
      float n0 = rrow[lane]      - qrow[lane];
      float n1 = rrow[lane + 32] - qrow[lane + 32];
      rrow[lane]      = n0;
      rrow[lane + 32] = n1;
      lsum = fmaf(n0, n0, fmaf(n1, n1, lsum));    // (quant - resid)^2 == new_resid^2
    }
  }
  #pragma unroll
  for (int s = 16; s >= 1; s >>= 1) lsum += __shfl_xor(lsum, s, 32);
  if (lane == 0) atomicAdd(lacc, lsum);

  // ---------------- Phase C: quant = x_proj - residual_final -> bf16 A-layout
  #pragma unroll
  for (int nt = 0; nt < 4; ++nt)
    #pragma unroll
    for (int r = 0; r < 8; ++r) {
      int row = hi * 8 + r, col = nt * 16 + m;
      qs[row * 68 + col] = f2bf(acc[nt][r] - rs[row * 68 + col]);
    }
  __syncthreads();
  Frag a2[2];                                     // A frags reused across all N chunks
  #pragma unroll
  for (int kk = 0; kk < 2; ++kk)
    #pragma unroll
    for (int v = 0; v < 8; ++v)
      a2[kk].u[v] = *(const uint32_t*)(qs + m * 68 + kk * 32 + kbase(v, hi));

  // ---------------- Phase D: z_q = quant @ W_out + b_out (N chunks of 128)
  for (int nc = 0; nc < DIN; nc += NC) {
    #pragma unroll
    for (int it = 0; it < 16; ++it) {
      int e = (it * BLOCK_T + tid) << 2;          // 8192 f32 per chunk
      int k = e >> 7, c = e & 127;
      const float4 w4 = *(const float4*)(Wout + (size_t)k * DIN + nc + c);
      wT[(c + 0) * 66 + k] = f2bf(w4.x);
      wT[(c + 1) * 66 + k] = f2bf(w4.y);
      wT[(c + 2) * 66 + k] = f2bf(w4.z);
      wT[(c + 3) * 66 + k] = f2bf(w4.w);
    }
    __syncthreads();
    #pragma unroll
    for (int nt = 0; nt < 8; ++nt) {
      v8f c2 = {};
      #pragma unroll
      for (int kk = 0; kk < 2; ++kk) {
        Frag b;
        #pragma unroll
        for (int v = 0; v < 8; ++v)
          b.u[v] = *(const uint32_t*)(wT + (nt * 16 + m) * 66 + kk * 32 + kbase(v, hi));
        c2 = __builtin_amdgcn_wmma_f32_16x16x32_bf16(
            false, a2[kk].v, false, b.v, (short)0, c2, false, false);
      }
      const int col  = nc + nt * 16 + m;
      const float bo = bout[col];
      const size_t row0 = wtok0 + (size_t)hi * 8;
      #pragma unroll
      for (int r = 0; r < 8; ++r)
        zq[(row0 + r) * DIN + col] = c2[r] + bo;  // lanes 0..15 coalesce over col
    }
    __syncthreads();
  }
}

extern "C" void kernel_launch(void* const* d_in, const int* in_sizes, int n_in,
                              void* d_out, int out_size, void* d_ws, size_t ws_size,
                              hipStream_t stream) {
  const float* x    = (const float*)d_in[0];
  const float* Win  = (const float*)d_in[1];
  const float* bin  = (const float*)d_in[2];
  const float* cbs  = (const float*)d_in[3];
  const float* Wout = (const float*)d_in[4];
  const float* bout = (const float*)d_in[5];

  float* out   = (float*)d_out;
  float* zq    = out;                               // 32*2048*1024
  float* idxf  = out + (size_t)67108864;            // 4*32*2048 indices (as float)
  float* lossp = out + (size_t)67371008;            // scalar vq_loss
  float* lacc  = (float*)d_ws;                      // loss accumulator

  rvq_zero<<<1, 1, 0, stream>>>(lacc);
  rvq_fused<<<TOK_TOTAL / 64, BLOCK_T, 0, stream>>>(
      x, Win, bin, cbs, Wout, bout, zq, idxf, lacc);
  rvq_finish<<<1, 1, 0, stream>>>(lacc, lossp);
}
